// STContrastiveReIDLoss_7499012899522
// MI455X (gfx1250) — matrix operand order repaired and checked by reading, not compiled, not run on hardware
//
#include <hip/hip_runtime.h>

// ---------------------------------------------------------------------------
// STContrastiveReIDLoss fused kernels for gfx1250 (MI455X, wave32, WMMA).
//
// Dominant cost: Gram = F F^T (8192x8192x2048 ~ 275 GFLOP) done as 3x bf16
// WMMA (hi*hi + hi*lo + lo*hi split) for ~fp32 accuracy at bf16 matrix-core
// rate. Features (64MB) live in L2, so this is compute-bound on WMMA.
// All BxB intermediates are reduced in-register in the GEMM epilogue; nothing
// BxB ever touches memory.
// Global->LDS staging uses GLOBAL_LOAD_ASYNC_TO_LDS_B128 (ASYNCcnt) when the
// toolchain declares the builtin, avoiding the VGPR round-trip; otherwise it
// falls back to load+ds_store.
// Workspace use: ~73.7 MB (bf16 F hi/lo 64MB, W hi/lo 6MB, row stats).
// ---------------------------------------------------------------------------

#define BN 8192
#define DK 2048
#define CN 751
#define CPAD 768
#define INV_TEMP 14.285714285714286f
#define MARGIN 0.3f

#if __has_builtin(__builtin_amdgcn_global_load_async_to_lds_b128)
#define HAVE_ASYNC_LDS 1
#else
#define HAVE_ASYNC_LDS 0
#endif

typedef __attribute__((ext_vector_type(16))) __bf16 bf16x16;
typedef __attribute__((ext_vector_type(8)))  float  f32x8;
typedef int v4i_t __attribute__((vector_size(16)));

// 16-byte global -> LDS copy. Async (no VGPR round-trip) when available.
__device__ inline void cp16_g2l(const __bf16* g, __bf16* l) {
#if HAVE_ASYNC_LDS
  // Builtin signature (from toolchain diagnostic): param0 is AS1 v4i*,
  // param1 LDS-side pointer; use real addrspacecasts (generic->AS1/AS3).
  __builtin_amdgcn_global_load_async_to_lds_b128(
      (__attribute__((address_space(1))) v4i_t*)(v4i_t*)g,
      (__attribute__((address_space(3))) v4i_t*)(v4i_t*)l,
      0, 0);
#else
  *reinterpret_cast<float4*>(l) = *reinterpret_cast<const float4*>(g);
#endif
}

// Fence this thread's outstanding async LDS writes (no-op on sync path).
__device__ inline void cp_fence() {
#if HAVE_ASYNC_LDS
#if __has_builtin(__builtin_amdgcn_s_wait_asynccnt)
  __builtin_amdgcn_s_wait_asynccnt(0);
#else
  asm volatile("s_wait_asynccnt 0x0" ::: "memory");
#endif
#endif
}

// A-matrix 16x32 bf16 fragment (ISA 7.12.2): lane L (h=L/16) holds
// K = 8h+0..7 (elems 0..7) and K = 16+8h+0..7 (elems 8..15), row M = L%16.
__device__ inline bf16x16 frag_ld_a(const __bf16* rowp, int h) {
  union { float4 q[2]; bf16x16 v; } u;
  const float4* b = reinterpret_cast<const float4*>(rowp);
  u.q[0] = b[h];      // k = 8h .. 8h+7
  u.q[1] = b[2 + h];  // k = 16+8h .. 16+8h+7
  return u.v;
}
// B-matrix 32x16 bf16 fragment: lane L (N = L%16, h = L/16) holds
// K = 8h+0..7 (elems 0..7) and K = 16+8h+0..7 (elems 8..15), matching A.
__device__ inline bf16x16 frag_ld_b(const __bf16* rowp, int h) {
  union { float4 q[2]; bf16x16 v; } u;
  const float4* b = reinterpret_cast<const float4*>(rowp);
  u.q[0] = b[h];
  u.q[1] = b[2 + h];
  return u.v;
}

// G ~= Ahi*Bhi + Ahi*Blo + Alo*Bhi  (3xBF16 split, ~fp32 accuracy)
__device__ inline f32x8 wmma3(bf16x16 ah, bf16x16 al, bf16x16 bh, bf16x16 bl,
                              f32x8 c) {
  c = __builtin_amdgcn_wmma_f32_16x16x32_bf16(false, ah, false, bh, (short)0, c, false, false);
  c = __builtin_amdgcn_wmma_f32_16x16x32_bf16(false, ah, false, bl, (short)0, c, false, false);
  c = __builtin_amdgcn_wmma_f32_16x16x32_bf16(false, al, false, bh, (short)0, c, false, false);
  return c;
}

// ---------------------------------------------------------------------------
// Row prep: sq[i] = |f_i|^2, inn[i] = 1/|f_i|, bf16 hi/lo split of features.
// One block (256 threads) per row; each thread handles 8 contiguous elements.
// ---------------------------------------------------------------------------
__global__ __launch_bounds__(256) void k_rowprep(const float* __restrict__ F,
                                                 __bf16* __restrict__ Fhi,
                                                 __bf16* __restrict__ Flo,
                                                 float* __restrict__ sq,
                                                 float* __restrict__ inn) {
  const int i = blockIdx.x, t = threadIdx.x;
  const size_t base = (size_t)i * DK + t * 8;
  float4 x0 = reinterpret_cast<const float4*>(F + base)[0];
  float4 x1 = reinterpret_cast<const float4*>(F + base)[1];
  float xs[8] = {x0.x, x0.y, x0.z, x0.w, x1.x, x1.y, x1.z, x1.w};
  union Pk { __bf16 hx[8]; float4 q; } ph, pl;
  float acc = 0.f;
#pragma unroll
  for (int e = 0; e < 8; ++e) {
    float x = xs[e];
    acc += x * x;
    __bf16 hb = (__bf16)x;
    ph.hx[e] = hb;
    pl.hx[e] = (__bf16)(x - (float)hb);
  }
  *reinterpret_cast<float4*>(Fhi + base) = ph.q;
  *reinterpret_cast<float4*>(Flo + base) = pl.q;

  __shared__ float red[256];
  red[t] = acc;
  __syncthreads();
  for (int s2 = 128; s2 > 0; s2 >>= 1) {
    if (t < s2) red[t] += red[t + s2];
    __syncthreads();
  }
  if (t == 0) {
    float v = red[0];
    sq[i] = v;
    inn[i] = 1.0f / sqrtf(v);
  }
}

// W (751x2048) -> bf16 hi/lo padded to 768 rows (pad rows = 0).
__global__ __launch_bounds__(256) void k_wprep(const float* __restrict__ W,
                                               __bf16* __restrict__ Whi,
                                               __bf16* __restrict__ Wlo) {
  const int c = blockIdx.x, t = threadIdx.x;
  const size_t dstb = (size_t)c * DK + t * 8;
  union Pk { __bf16 hx[8]; float4 q; } ph, pl;
  if (c < CN) {
    const size_t srcb = (size_t)c * DK + t * 8;
    float4 x0 = reinterpret_cast<const float4*>(W + srcb)[0];
    float4 x1 = reinterpret_cast<const float4*>(W + srcb)[1];
    float xs[8] = {x0.x, x0.y, x0.z, x0.w, x1.x, x1.y, x1.z, x1.w};
#pragma unroll
    for (int e = 0; e < 8; ++e) {
      __bf16 hb = (__bf16)xs[e];
      ph.hx[e] = hb;
      pl.hx[e] = (__bf16)(xs[e] - (float)hb);
    }
  } else {
    float4 z{0.f, 0.f, 0.f, 0.f};
    ph.q = z;
    pl.q = z;
  }
  *reinterpret_cast<float4*>(Whi + dstb) = ph.q;
  *reinterpret_cast<float4*>(Wlo + dstb) = pl.q;
}

// Zero row-stat arrays + scalar accumulators.
__global__ void k_zero(int* hpI, int* hnI, float* sS, float* psS, float* npS,
                       int* paI, float* accv) {
  int i = blockIdx.x * blockDim.x + threadIdx.x;
  if (i < BN) {
    hpI[i] = 0;                        // 0.0f bits (d2 >= 0)
    hnI[i] = __float_as_int(1.0e9f);   // BIG
    sS[i] = 0.f;
    psS[i] = 0.f;
    npS[i] = 0.f;
    paI[i] = 0;
  }
  if (i < 8) accv[i] = 0.f;
}

// ---------------------------------------------------------------------------
// ID loss: logits = F W^T + b, fused per-row online logsumexp + label logit.
// Grid: 64 blocks of 128 rows. 8 waves, wave w owns rows [w*16, w*16+16).
// ---------------------------------------------------------------------------
__global__ __launch_bounds__(256) void k_id(const __bf16* __restrict__ Fhi,
                                            const __bf16* __restrict__ Flo,
                                            const __bf16* __restrict__ Whi,
                                            const __bf16* __restrict__ Wlo,
                                            const float* __restrict__ bias,
                                            const int* __restrict__ labels,
                                            float* accv) {
  __shared__ __bf16 sAhi[128 * 32], sAlo[128 * 32], sBhi[128 * 32], sBlo[128 * 32];
  const int tid = threadIdx.x, lane = tid & 31, wave = tid >> 5;
  const int h = lane >> 4, n16 = lane & 15;
  const int I0 = blockIdx.x * 128;
  const int ibase = I0 + wave * 16;

  int labI[8];
#pragma unroll
  for (int r = 0; r < 8; ++r) labI[r] = labels[ibase + r + 8 * h];

  float m[8], s[8], lv[8];
#pragma unroll
  for (int r = 0; r < 8; ++r) { m[r] = -1e30f; s[r] = 0.f; lv[r] = 0.f; }

  for (int jb = 0; jb < 6; ++jb) {
    const int J0 = jb * 128;
    f32x8 acc[8];
    f32x8 z = {0, 0, 0, 0, 0, 0, 0, 0};
#pragma unroll
    for (int jt = 0; jt < 8; ++jt) acc[jt] = z;

    for (int kc = 0; kc < DK; kc += 32) {
      __syncthreads();  // prior fragment ds_loads done before LDS overwrite
      for (int t = tid; t < 512; t += 256) {
        int row = t >> 2, seg = t & 3;
        size_t ga = (size_t)(I0 + row) * DK + kc + seg * 8;
        size_t gb = (size_t)(J0 + row) * DK + kc + seg * 8;
        int lo = row * 32 + seg * 8;
        cp16_g2l(Fhi + ga, sAhi + lo);
        cp16_g2l(Flo + ga, sAlo + lo);
        cp16_g2l(Whi + gb, sBhi + lo);
        cp16_g2l(Wlo + gb, sBlo + lo);
      }
      cp_fence();
      __syncthreads();
      bf16x16 aH = frag_ld_a(sAhi + (wave * 16 + n16) * 32, h);
      bf16x16 aL = frag_ld_a(sAlo + (wave * 16 + n16) * 32, h);
#pragma unroll
      for (int jt = 0; jt < 8; ++jt) {
        bf16x16 bH = frag_ld_b(sBhi + (jt * 16 + n16) * 32, h);
        bf16x16 bL = frag_ld_b(sBlo + (jt * 16 + n16) * 32, h);
        acc[jt] = wmma3(aH, aL, bH, bL, acc[jt]);
      }
    }
    // epilogue: online softmax over classes
#pragma unroll
    for (int jt = 0; jt < 8; ++jt) {
      int j = J0 + jt * 16 + n16;
      bool vj = (j < CN);
      int jc = vj ? j : (CN - 1);
      float bj = bias[jc];
#pragma unroll
      for (int r = 0; r < 8; ++r) {
        float v = vj ? (acc[jt][r] + bj) : -1e30f;
        float nm = fmaxf(m[r], v);
        s[r] = s[r] * __expf(m[r] - nm) + __expf(v - nm);
        m[r] = nm;
        if (vj && j == labI[r]) lv[r] = v;
      }
    }
  }
  // merge (m,s) and label logit across the 16 lanes sharing each row
#pragma unroll
  for (int r = 0; r < 8; ++r) {
    float mm = m[r], ss = s[r], ll = lv[r];
#pragma unroll
    for (int mk = 1; mk <= 8; mk <<= 1) {
      float om = __shfl_xor(mm, mk, 32);
      float os = __shfl_xor(ss, mk, 32);
      float nm = fmaxf(mm, om);
      ss = ss * __expf(mm - nm) + os * __expf(om - nm);
      mm = nm;
      ll += __shfl_xor(ll, mk, 32);
    }
    if (n16 == 0) atomicAdd(accv + 0, (mm + __logf(ss)) - ll);
  }
}

// ---------------------------------------------------------------------------
// Gram + fused triplet / spatio-temporal stats.
// Grid (64, 8): block.x = 128-row anchor block, block.y = 1/8th of columns.
// Per row i (streamed over j): hardest-pos max(d2), hardest-neg min(d2)
// (diagonal is in neg-mask, matching reference), softmax sum with exact shift
// 1/TEMP (sim_ii == 1/TEMP is the row max by Cauchy-Schwarz), ST-positive
// sim sum + count. Merged via shfl butterflies then per-row global atomics.
// ---------------------------------------------------------------------------
__global__ __launch_bounds__(256) void k_gram(const __bf16* __restrict__ Fhi,
                                              const __bf16* __restrict__ Flo,
                                              const int* __restrict__ labels,
                                              const int* __restrict__ cams,
                                              const float* __restrict__ ts,
                                              const float* __restrict__ reach,
                                              const float* __restrict__ sqv,
                                              const float* __restrict__ innv,
                                              int* hpI, int* hnI, float* sS,
                                              float* psS, float* npS, int* paI) {
  __shared__ __bf16 sAhi[128 * 32], sAlo[128 * 32], sBhi[128 * 32], sBlo[128 * 32];
  const int tid = threadIdx.x, lane = tid & 31, wave = tid >> 5;
  const int h = lane >> 4, n16 = lane & 15;
  const int I0 = blockIdx.x * 128;
  const int ibase = I0 + wave * 16;

  int labI[8], camI[8];
  float tI[8], sqI[8], inI[8];
#pragma unroll
  for (int r = 0; r < 8; ++r) {
    int i = ibase + r + 8 * h;
    labI[r] = labels[i];
    camI[r] = cams[i];
    tI[r] = ts[i];
    sqI[r] = sqv[i];
    inI[r] = innv[i];
  }

  float hpv[8], hnv[8], sv[8], psv[8], npv[8], pav[8];
#pragma unroll
  for (int r = 0; r < 8; ++r) {
    hpv[r] = 0.f; hnv[r] = 1.0e9f; sv[r] = 0.f;
    psv[r] = 0.f; npv[r] = 0.f;   pav[r] = 0.f;
  }

  for (int jb = blockIdx.y * 8; jb < blockIdx.y * 8 + 8; ++jb) {
    const int J0 = jb * 128;
    f32x8 acc[8];
    f32x8 z = {0, 0, 0, 0, 0, 0, 0, 0};
#pragma unroll
    for (int jt = 0; jt < 8; ++jt) acc[jt] = z;

    for (int kc = 0; kc < DK; kc += 32) {
      __syncthreads();
      for (int t = tid; t < 512; t += 256) {
        int row = t >> 2, seg = t & 3;
        size_t ga = (size_t)(I0 + row) * DK + kc + seg * 8;
        size_t gb = (size_t)(J0 + row) * DK + kc + seg * 8;
        int lo = row * 32 + seg * 8;
        cp16_g2l(Fhi + ga, sAhi + lo);
        cp16_g2l(Flo + ga, sAlo + lo);
        cp16_g2l(Fhi + gb, sBhi + lo);
        cp16_g2l(Flo + gb, sBlo + lo);
      }
      cp_fence();
      __syncthreads();
      bf16x16 aH = frag_ld_a(sAhi + (wave * 16 + n16) * 32, h);
      bf16x16 aL = frag_ld_a(sAlo + (wave * 16 + n16) * 32, h);
#pragma unroll
      for (int jt = 0; jt < 8; ++jt) {
        bf16x16 bH = frag_ld_b(sBhi + (jt * 16 + n16) * 32, h);
        bf16x16 bL = frag_ld_b(sBlo + (jt * 16 + n16) * 32, h);
        acc[jt] = wmma3(aH, aL, bH, bL, acc[jt]);
      }
    }
    // fused epilogue on the 128x128 tile block
#pragma unroll
    for (int jt = 0; jt < 8; ++jt) {
      int j = J0 + jt * 16 + n16;
      int labJ = labels[j];
      int camJ = cams[j];
      float tJ = ts[j], sqJ = sqv[j], inJ = innv[j];
#pragma unroll
      for (int r = 0; r < 8; ++r) {
        int i = ibase + r + 8 * h;
        float G = acc[jt][r];
        float d2 = fmaxf(sqI[r] + sqJ - 2.0f * G, 0.0f);
        bool same = (labI[r] == labJ);
        bool eye = (i == j);
        bool pos = same && !eye;
        if (pos) { hpv[r] = fmaxf(hpv[r], d2); pav[r] = 1.f; }
        else     { hnv[r] = fminf(hnv[r], d2); }
        float sim = G * inI[r] * inJ * INV_TEMP;
        sv[r] += __expf(sim - INV_TEMP);  // row max is the diagonal = 1/TEMP
        float thr = reach[camI[r] * 16 + camJ];
        if (pos && fabsf(tI[r] - tJ) <= thr) { psv[r] += sim; npv[r] += 1.f; }
      }
    }
  }

  // reduce across the 16 lanes holding each row, then per-row atomics
#pragma unroll
  for (int r = 0; r < 8; ++r) {
    float hp = hpv[r], hn = hnv[r], ss = sv[r], pp = psv[r], nn = npv[r], pa = pav[r];
#pragma unroll
    for (int mk = 1; mk <= 8; mk <<= 1) {
      hp = fmaxf(hp, __shfl_xor(hp, mk, 32));
      hn = fminf(hn, __shfl_xor(hn, mk, 32));
      ss += __shfl_xor(ss, mk, 32);
      pp += __shfl_xor(pp, mk, 32);
      nn += __shfl_xor(nn, mk, 32);
      pa = fmaxf(pa, __shfl_xor(pa, mk, 32));
    }
    if (n16 == 0) {
      int i = ibase + r + 8 * h;
      atomicMax(hpI + i, __float_as_int(hp));  // valid: hp >= 0
      atomicMin(hnI + i, __float_as_int(hn));  // valid: hn >= 0
      atomicAdd(sS + i, ss);
      atomicAdd(psS + i, pp);
      atomicAdd(npS + i, nn);
      if (pa > 0.5f) atomicOr(paI + i, 1);
    }
  }
}

// Per-row finalize -> scalar accumulators.
__global__ __launch_bounds__(256) void k_rows(const int* hpI, const int* hnI,
                                              const float* sS, const float* psS,
                                              const float* npS, const int* paI,
                                              float* accv) {
  int i = blockIdx.x * 256 + threadIdx.x;
  float hp = __int_as_float(hpI[i]);
  float hn = __int_as_float(hnI[i]);
  int pa = paI[i];
  // neg always exists (diagonal is in neg_mask) -> valid == any(pos)
  float tri = pa ? fmaxf(hp - hn + MARGIN, 0.f) : 0.f;
  float nv = pa ? 1.f : 0.f;
  float lse = INV_TEMP + __logf(sS[i]);
  float st = psS[i] - npS[i] * lse;  // sum over st_pos of logp_sim (row i)
  float np = npS[i];
#pragma unroll
  for (int mk = 1; mk <= 16; mk <<= 1) {
    tri += __shfl_xor(tri, mk, 32);
    nv += __shfl_xor(nv, mk, 32);
    st += __shfl_xor(st, mk, 32);
    np += __shfl_xor(np, mk, 32);
  }
  if ((threadIdx.x & 31) == 0) {
    atomicAdd(accv + 1, tri);
    atomicAdd(accv + 2, nv);
    atomicAdd(accv + 3, st);
    atomicAdd(accv + 4, np);
  }
}

__global__ void k_final(const float* accv, float* out) {
  float lid = accv[0] * (1.0f / (float)BN);
  float ltri = (accv[2] > 0.f) ? (accv[1] / fmaxf(accv[2], 1.f)) : 0.f;
  float lst = (accv[4] > 0.f) ? (-accv[3] / fmaxf(accv[4], 1.f)) : 0.f;
  out[0] = lid + 0.5f * ltri + 0.3f * lst;
}

// ---------------------------------------------------------------------------
// Workspace layout (bytes). Total ~73.7 MB.
// ---------------------------------------------------------------------------
#define OFF_FHI 0ull
#define OFF_FLO 33554432ull
#define OFF_WHI 67108864ull
#define OFF_WLO 70254592ull
#define OFF_SQ  73400320ull
#define OFF_INN 73433088ull
#define OFF_HP  73465856ull
#define OFF_HN  73498624ull
#define OFF_S   73531392ull
#define OFF_PS  73564160ull
#define OFF_NP  73596928ull
#define OFF_PA  73629696ull
#define OFF_ACC 73662464ull

extern "C" void kernel_launch(void* const* d_in, const int* in_sizes, int n_in,
                              void* d_out, int out_size, void* d_ws, size_t ws_size,
                              hipStream_t stream) {
  const float* features = (const float*)d_in[0];
  const int* labels     = (const int*)d_in[1];
  const int* cams       = (const int*)d_in[2];
  const float* tsmp     = (const float*)d_in[3];
  const float* reach    = (const float*)d_in[4];
  const float* W        = (const float*)d_in[5];
  const float* bias     = (const float*)d_in[6];
  float* out            = (float*)d_out;

  char* ws = (char*)d_ws;
  __bf16* Fhi = (__bf16*)(ws + OFF_FHI);
  __bf16* Flo = (__bf16*)(ws + OFF_FLO);
  __bf16* Whi = (__bf16*)(ws + OFF_WHI);
  __bf16* Wlo = (__bf16*)(ws + OFF_WLO);
  float* sq   = (float*)(ws + OFF_SQ);
  float* inn  = (float*)(ws + OFF_INN);
  int* hpI    = (int*)(ws + OFF_HP);
  int* hnI    = (int*)(ws + OFF_HN);
  float* sS   = (float*)(ws + OFF_S);
  float* psS  = (float*)(ws + OFF_PS);
  float* npS  = (float*)(ws + OFF_NP);
  int* paI    = (int*)(ws + OFF_PA);
  float* accv = (float*)(ws + OFF_ACC);

  k_zero<<<(BN + 255) / 256, 256, 0, stream>>>(hpI, hnI, sS, psS, npS, paI, accv);
  k_rowprep<<<BN, 256, 0, stream>>>(features, Fhi, Flo, sq, inn);
  k_wprep<<<CPAD, 256, 0, stream>>>(W, Whi, Wlo);
  k_id<<<BN / 128, 256, 0, stream>>>(Fhi, Flo, Whi, Wlo, bias, labels, accv);
  dim3 ggrid(BN / 128, 8, 1);
  k_gram<<<ggrid, 256, 0, stream>>>(Fhi, Flo, labels, cams, tsmp, reach, sq, inn,
                                    hpI, hnI, sS, psS, npS, paI);
  k_rows<<<BN / 256, 256, 0, stream>>>(hpI, hnI, sS, psS, npS, paI, accv);
  k_final<<<1, 1, 0, stream>>>(accv, out);
}
